// StyleInjector_30176440222298
// MI455X (gfx1250) — matrix operand order repaired and checked by reading, not compiled, not run on hardware
//
#include <hip/hip_runtime.h>

typedef __attribute__((ext_vector_type(16))) _Float16 v16h;
typedef __attribute__((ext_vector_type(8)))  _Float16 v8h;
typedef __attribute__((ext_vector_type(4)))  _Float16 v4h;
typedef __attribute__((ext_vector_type(8)))  float    v8f;

#define BATCH 8
#define LSEQ  8192
#define CDIM  1024
#define WDIM  512
#define LDST  40   // padded LDS row stride in f16 (80B -> conflict-free b128)

// ---------------- prep kernel 1: styles = w @ (affine_w*gain)^T + affine_b ----
__global__ __launch_bounds__(256) void k_styles(const float* __restrict__ w,
                                                const float* __restrict__ aw,
                                                const float* __restrict__ ab,
                                                float* __restrict__ styles) {
  const int o = blockIdx.x * 256 + threadIdx.x;   // 0..1023
  const int b = blockIdx.y;                       // 0..7
  const float gain = 0.04419417382415922f;        // 1/sqrt(512)
  const float* wr = w + b * WDIM;
  const float* ar = aw + (size_t)o * WDIM;
  float acc = 0.f;
#pragma unroll 8
  for (int k = 0; k < WDIM; ++k) acc = fmaf(wr[k], ar[k], acc);
  styles[b * CDIM + o] = fmaf(acc, gain, ab[o]);
}

// ---------------- prep kernel 2: dcoefs = rsqrt(sum_c (weight*styles)^2) ------
__global__ __launch_bounds__(256) void k_dcoefs(const float* __restrict__ weight,
                                                const float* __restrict__ styles,
                                                float* __restrict__ dcoefs) {
  const int o = blockIdx.x * 256 + threadIdx.x;
  const int b = blockIdx.y;
  const float4* wr = (const float4*)(weight + (size_t)o * CDIM);
  const float4* sr = (const float4*)(styles + b * CDIM);
  float acc = 0.f;
#pragma unroll 4
  for (int c = 0; c < CDIM / 4; ++c) {
    float4 wv = wr[c], sv = sr[c];
    float t0 = wv.x * sv.x, t1 = wv.y * sv.y;
    float t2 = wv.z * sv.z, t3 = wv.w * sv.w;
    acc = fmaf(t0, t0, fmaf(t1, t1, fmaf(t2, t2, fmaf(t3, t3, acc))));
  }
  dcoefs[b * CDIM + o] = rsqrtf(acc + 1e-8f);
}

// ---------------- prep kernel 3: weight f32 -> f16 ----------------------------
__global__ __launch_bounds__(256) void k_wcvt(const float* __restrict__ weight,
                                              _Float16* __restrict__ wf16) {
  const int i = blockIdx.x * 256 + threadIdx.x;   // quad index
  float4 v = ((const float4*)weight)[i];
  v4h h;
  h[0] = (_Float16)v.x; h[1] = (_Float16)v.y;
  h[2] = (_Float16)v.z; h[3] = (_Float16)v.w;
  ((v4h*)wf16)[i] = h;
}

// ---------------- main kernel: modulated GEMM + demod + bias + clamp ---------
__global__ __launch_bounds__(256) void k_modgemm(
    const float* __restrict__ x, const _Float16* __restrict__ wf16,
    const float* __restrict__ styles, const float* __restrict__ dcoefs,
    const float* __restrict__ bias, float* __restrict__ out) {
  __shared__ _Float16 As[2][128 * LDST];   // double-buffered 128x32 f16 x-tile

  const int tid  = threadIdx.x;
  const int lane = tid & 31;
  const int wave = tid >> 5;
  const int wm   = wave & 3;               // 4 M-groups (32 rows each)
  const int wn   = wave >> 2;              // 2 N-groups (64 cols each)
  const int nblk = blockIdx.x;             // 8   (adjacent blocks share x tile)
  const int mblk = blockIdx.y;             // 64
  const int b    = blockIdx.z;             // 8
  const int mBase = mblk * 128;
  const int nBase = nblk * 128;

  const float* xblk = x + ((size_t)b * LSEQ + mBase) * CDIM;
  const float* srow = styles + b * CDIM;

  // staging: each thread stages rows {row0, row0+64}, K-chunk [kc, kc+8)
  const int row0 = tid >> 2;
  const int kc   = (tid & 3) * 8;

  // B fragment addressing: lane holds 16 contiguous c-values of one weight row
  const int bn  = lane & 15;
  const int bko = (lane >> 4) * 16;
  const _Float16* wrow0 = wf16 + (size_t)(nBase + wn * 64 + bn) * CDIM + bko;

  float dc[4], bi[4];
#pragma unroll
  for (int j = 0; j < 4; ++j) {
    const int n = nBase + wn * 64 + j * 16 + bn;
    dc[j] = dcoefs[b * CDIM + n];
    bi[j] = bias[n];
  }

  // A fragment addressing in LDS (16-bit 16x32 layout -> two b128 per lane)
  const int arow  = lane & 15;
  const int akoff = (lane >> 4) * 16;

  v8f acc[2][4];
#pragma unroll
  for (int i = 0; i < 2; ++i)
#pragma unroll
    for (int j = 0; j < 4; ++j)
#pragma unroll
      for (int e = 0; e < 8; ++e) acc[i][j][e] = 0.f;

  float4 g0[2], g1[2], s0, s1;

  // ---- prologue: stage k-block 0 into buffer 0 ----
  s0 = *(const float4*)(srow + kc);
  s1 = *(const float4*)(srow + kc + 4);
#pragma unroll
  for (int i = 0; i < 2; ++i) {
    const float* p = xblk + (size_t)(row0 + i * 64) * CDIM + kc;
    g0[i] = *(const float4*)p;
    g1[i] = *(const float4*)(p + 4);
  }
#pragma unroll
  for (int i = 0; i < 2; ++i) {
    v8h h;
    h[0] = (_Float16)(g0[i].x * s0.x); h[1] = (_Float16)(g0[i].y * s0.y);
    h[2] = (_Float16)(g0[i].z * s0.z); h[3] = (_Float16)(g0[i].w * s0.w);
    h[4] = (_Float16)(g1[i].x * s1.x); h[5] = (_Float16)(g1[i].y * s1.y);
    h[6] = (_Float16)(g1[i].z * s1.z); h[7] = (_Float16)(g1[i].w * s1.w);
    *(v8h*)(As[0] + (row0 + i * 64) * LDST + kc) = h;
  }
  __syncthreads();

  for (int kblk = 0; kblk < 32; ++kblk) {
    const int buf  = kblk & 1;
    const bool more = (kblk + 1) < 32;

    // software pipeline: issue global loads for next k-block early
    if (more) {
      const int kn = (kblk + 1) * 32 + kc;
      s0 = *(const float4*)(srow + kn);
      s1 = *(const float4*)(srow + kn + 4);
#pragma unroll
      for (int i = 0; i < 2; ++i) {
        const float* p = xblk + (size_t)(row0 + i * 64) * CDIM + kn;
        g0[i] = *(const float4*)p;
        g1[i] = *(const float4*)(p + 4);
      }
    }

    // A fragments from LDS
    v16h a[2];
#pragma unroll
    for (int i = 0; i < 2; ++i) {
      const _Float16* ap = As[buf] + (wm * 32 + i * 16 + arow) * LDST + akoff;
      v8h lo = *(const v8h*)ap;
      v8h hi = *(const v8h*)(ap + 8);
      a[i] = __builtin_shufflevector(lo, hi,
                                     0, 1, 2, 3, 4, 5, 6, 7,
                                     8, 9, 10, 11, 12, 13, 14, 15);
    }

    // B fragments direct from L2-resident f16 weight; 8 WMMAs per wave
#pragma unroll
    for (int j = 0; j < 4; ++j) {
      v16h bf = *(const v16h*)(wrow0 + (size_t)j * 16 * CDIM + kblk * 32);
#pragma unroll
      for (int i = 0; i < 2; ++i) {
        acc[i][j] = __builtin_amdgcn_wmma_f32_16x16x32_f16(
            false, a[i], false, bf, (short)0, acc[i][j], false, false);
      }
    }

    // convert + store next tile into the other buffer
    if (more) {
#pragma unroll
      for (int i = 0; i < 2; ++i) {
        v8h h;
        h[0] = (_Float16)(g0[i].x * s0.x); h[1] = (_Float16)(g0[i].y * s0.y);
        h[2] = (_Float16)(g0[i].z * s0.z); h[3] = (_Float16)(g0[i].w * s0.w);
        h[4] = (_Float16)(g1[i].x * s1.x); h[5] = (_Float16)(g1[i].y * s1.y);
        h[6] = (_Float16)(g1[i].z * s1.z); h[7] = (_Float16)(g1[i].w * s1.w);
        *(v8h*)(As[buf ^ 1] + (row0 + i * 64) * LDST + kc) = h;
      }
    }
    __syncthreads();
  }

  // ---- epilogue: demodulate, bias, clamp, store (C/D layout: M=r+8*(lane/16), N=lane%16)
#pragma unroll
  for (int i = 0; i < 2; ++i) {
#pragma unroll
    for (int j = 0; j < 4; ++j) {
      const int n = nBase + wn * 64 + j * 16 + bn;
#pragma unroll
      for (int r = 0; r < 8; ++r) {
        const int m = mBase + wm * 32 + i * 16 + ((lane >> 4) * 8) + r;
        float v = fmaf(acc[i][j][r], dc[j], bi[j]);
        v = fminf(fmaxf(v, -256.0f), 256.0f);
        out[((size_t)b * LSEQ + m) * CDIM + n] = v;
      }
    }
  }
}

extern "C" void kernel_launch(void* const* d_in, const int* in_sizes, int n_in,
                              void* d_out, int out_size, void* d_ws, size_t ws_size,
                              hipStream_t stream) {
  const float* x    = (const float*)d_in[0];   // [8, 8192, 1024]
  const float* w    = (const float*)d_in[1];   // [8, 512]
  const float* wt   = (const float*)d_in[2];   // [1024, 1024]
  const float* bias = (const float*)d_in[3];   // [1024]
  const float* aw   = (const float*)d_in[4];   // [1024, 512]
  const float* ab   = (const float*)d_in[5];   // [1024]
  float* out = (float*)d_out;

  char* ws = (char*)d_ws;
  float*    styles = (float*)ws;                 // 32 KB
  float*    dcoefs = (float*)(ws + 32768);       // 32 KB
  _Float16* wf16   = (_Float16*)(ws + 65536);    // 2 MB

  k_styles<<<dim3(CDIM / 256, BATCH), 256, 0, stream>>>(w, aw, ab, styles);
  k_wcvt<<<dim3((CDIM * CDIM / 4) / 256), 256, 0, stream>>>(wt, wf16);
  k_dcoefs<<<dim3(CDIM / 256, BATCH), 256, 0, stream>>>(wt, styles, dcoefs);
  k_modgemm<<<dim3(CDIM / 128, LSEQ / 128, BATCH), 256, 0, stream>>>(
      x, wf16, styles, dcoefs, bias, out);
}
